// SparseMoE_81922206204119
// MI455X (gfx1250) — compile-verified, matrix operand order checked by reading, not run on hardware
//
#include <hip/hip_runtime.h>
#include <cstdint>
#include <cstddef>

#define NE    8
#define TOPK  2
#define NTOK  16384       // B*T = 8*2048
#define DIM   512
#define HID   1408
#define TM    64          // gathered tokens per block tile
#define TH    128         // H chunk (8 waves x 16 cols)

typedef unsigned short u16;
typedef __attribute__((ext_vector_type(16))) __bf16 v16bf;
typedef __attribute__((ext_vector_type(8)))  float  v8f;
typedef __attribute__((ext_vector_type(8)))  unsigned short v8u16;

#if __has_builtin(__builtin_amdgcn_global_load_async_to_lds_b128) && \
    __has_builtin(__builtin_amdgcn_s_wait_asynccnt)
#define HAVE_ASYNC_LDS 1
typedef __attribute__((ext_vector_type(4))) int v4i_t;
typedef __attribute__((address_space(1))) v4i_t* gv4p;   // global int4*
typedef __attribute__((address_space(3))) v4i_t* lv4p;   // LDS int4*
#else
#define HAVE_ASYNC_LDS 0
#endif

static __device__ __forceinline__ u16 f2bf(float f) {
  unsigned u = __float_as_uint(f);
  unsigned r = u + 0x7FFFu + ((u >> 16) & 1u);   // round-to-nearest-even
  return (u16)(r >> 16);
}

// Build a 16x32-bf16 WMMA fragment from two contiguous 8-element (16B) chunks.
static __device__ __forceinline__ v16bf load_frag(const u16* p0, const u16* p1) {
  union { v16bf v; v8u16 h[2]; } u;
  u.h[0] = *(const v8u16*)p0;   // K offs +0..7   -> VGPR 0..3
  u.h[1] = *(const v8u16*)p1;   // K offs +16..23 -> VGPR 4..7
  return u.v;
}

static __device__ __forceinline__ v8f wmma_bf16(v16bf a, v16bf b, v8f c) {
  return __builtin_amdgcn_wmma_f32_16x16x32_bf16(false, a, false, b,
                                                 (short)0, c, false, false);
}

// ---------------------------------------------------------------- converts
__global__ __launch_bounds__(256)
void f32_to_bf16_kernel(const float* __restrict__ in, u16* __restrict__ out, int n8) {
  int i = blockIdx.x * 256 + threadIdx.x;
  if (i >= n8) return;
  const float4* ip = (const float4*)in;
  float4 a = ip[2 * i], b = ip[2 * i + 1];
  union { v8u16 v; u16 s[8]; } o;
  o.s[0] = f2bf(a.x); o.s[1] = f2bf(a.y); o.s[2] = f2bf(a.z); o.s[3] = f2bf(a.w);
  o.s[4] = f2bf(b.x); o.s[5] = f2bf(b.y); o.s[6] = f2bf(b.z); o.s[7] = f2bf(b.w);
  ((v8u16*)out)[i] = o.v;
}

// ---------------------------------------------------------------- router
__global__ __launch_bounds__(256)
void router_kernel(const float* __restrict__ x, const float* __restrict__ rw,
                   int* __restrict__ topk_e, float* __restrict__ topk_g,
                   int* __restrict__ counts, float* __restrict__ usage) {
  __shared__ float rws[NE * DIM];   // 16 KB
  __shared__ float us[NE];
  int tid = threadIdx.x;
  for (int i = tid; i < NE * DIM; i += 256) rws[i] = rw[i];
  if (tid < NE) us[tid] = 0.f;
  __syncthreads();

  int n = blockIdx.x * 256 + tid;
  float logit[NE];
#pragma unroll
  for (int e = 0; e < NE; ++e) logit[e] = 0.f;
  const float4* xr = (const float4*)(x + (size_t)n * DIM);
  for (int c = 0; c < DIM / 4; ++c) {
    float4 xv = xr[c];
#pragma unroll
    for (int e = 0; e < NE; ++e) {
      float4 wv = ((const float4*)(rws + e * DIM))[c];
      logit[e] += xv.x * wv.x + xv.y * wv.y + xv.z * wv.z + xv.w * wv.w;
    }
  }
  float mx = logit[0];
#pragma unroll
  for (int e = 1; e < NE; ++e) mx = fmaxf(mx, logit[e]);
  float p[NE], sum = 0.f;
#pragma unroll
  for (int e = 0; e < NE; ++e) { p[e] = __expf(logit[e] - mx); sum += p[e]; }
  float inv = 1.f / sum;
#pragma unroll
  for (int e = 0; e < NE; ++e) p[e] *= inv;

  int i1 = 0; float b1v = p[0];
#pragma unroll
  for (int e = 1; e < NE; ++e) if (p[e] > b1v) { b1v = p[e]; i1 = e; }
  int i2 = -1; float b2v = -1.f;
#pragma unroll
  for (int e = 0; e < NE; ++e) if (e != i1 && p[e] > b2v) { b2v = p[e]; i2 = e; }
  float rs = 1.f / (b1v + b2v);

  topk_e[n * 2 + 0] = i1; topk_g[n * 2 + 0] = b1v * rs;
  topk_e[n * 2 + 1] = i2; topk_g[n * 2 + 1] = b2v * rs;
  atomicAdd(&counts[i1], 1);
  atomicAdd(&counts[i2], 1);
#pragma unroll
  for (int e = 0; e < NE; ++e) atomicAdd(&us[e], p[e]);
  __syncthreads();
  if (tid < NE) atomicAdd(&usage[tid], us[tid]);
}

// ---------------------------------------------------------------- scan + aux
__global__ void finalize_router_kernel(const int* __restrict__ counts,
                                       int* __restrict__ offs, int* __restrict__ cursor,
                                       const float* __restrict__ usage,
                                       float* __restrict__ aux_out) {
  if (threadIdx.x == 0) {
    int acc = 0;
    for (int e = 0; e < NE; ++e) { offs[e] = acc; cursor[e] = acc; acc += counts[e]; }
    offs[NE] = acc;
    float s = 0.f;
    for (int e = 0; e < NE; ++e) { float u = usage[e] * (1.f / NTOK); s += u * u; }
    aux_out[0] = (float)NE * s * 0.01f;
  }
}

// ---------------------------------------------------------------- scatter
__global__ __launch_bounds__(256)
void scatter_kernel(const int* __restrict__ topk_e, const float* __restrict__ topk_g,
                    int* __restrict__ cursor, int* __restrict__ gidx,
                    float* __restrict__ ggate) {
  int s = blockIdx.x * 256 + threadIdx.x;
  int e = topk_e[s];
  int pos = atomicAdd(&cursor[e], 1);
  gidx[pos]  = s >> 1;
  ggate[pos] = topk_g[s];
}

// ---------------------------------------------------------------- fused FFN
__global__ __launch_bounds__(256)
void moe_ffn_kernel(const u16* __restrict__ xb,  const u16* __restrict__ w1b,
                    const u16* __restrict__ w2b, const u16* __restrict__ w3b,
                    const int* __restrict__ gidx, const float* __restrict__ ggate,
                    const int* __restrict__ offs, float* __restrict__ out) {
  const int e = blockIdx.y;
  const int rowBeg = offs[e];
  const int rowEnd = offs[e + 1];
  const int row0 = rowBeg + blockIdx.x * TM;
  if (row0 >= rowEnd) return;
  const int nrows = min(TM, rowEnd - row0);

  __shared__ u16  xs[TM * DIM];   // 64 KB bf16 token tile
  __shared__ u16  hs[TM * TH];    // 16 KB bf16 h chunk
  __shared__ int  tok_s[TM];
  __shared__ float gate_s[TM];

  const int tid  = threadIdx.x;
  const int lane = tid & 31;
  const int wave = tid >> 5;          // 8 waves
  const int hsel = lane >> 4;         // fragment K-half select
  const int lr   = lane & 15;

  if (tid < TM) {
    int tok = -1; float g = 0.f;
    if (tid < nrows) { tok = gidx[row0 + tid]; g = ggate[row0 + tid]; }
    tok_s[tid] = tok; gate_s[tid] = g;
  }
  __syncthreads();

  // Gather 64 token rows of bf16 x into LDS (128-bit transfers).
  // Prefer the CDNA5 async memory->LDS path (no VGPR round-trip, ASYNCcnt).
  for (int v = tid; v < TM * (DIM / 8); v += 256) {
    int r = v >> 6;                  // DIM/8 == 64
    int c = v & 63;
    int tok = tok_s[r];
#if HAVE_ASYNC_LDS
    int tsrc = (tok >= 0) ? tok : 0; // dead rows: any data, gated off later
    __builtin_amdgcn_global_load_async_to_lds_b128(
        (gv4p)(xb + (size_t)tsrc * DIM + c * 8),
        (lv4p)(((uint4*)xs) + v), 0, 0);
#else
    uint4 val = make_uint4(0u, 0u, 0u, 0u);
    if (tok >= 0) val = ((const uint4*)(xb + (size_t)tok * DIM))[c];
    ((uint4*)xs)[v] = val;
#endif
  }
#if HAVE_ASYNC_LDS
  __builtin_amdgcn_s_wait_asynccnt(0);
#endif

  const u16* w1e = w1b + (size_t)e * HID * DIM;
  const u16* w3e = w3b + (size_t)e * HID * DIM;
  const u16* w2e = w2b + (size_t)e * DIM * HID;

  const v8f zf = {0.f, 0.f, 0.f, 0.f, 0.f, 0.f, 0.f, 0.f};
  v8f acc2[4][4];                    // 64 x 64 output per wave (wave owns 64 D-cols)
#pragma unroll
  for (int mt = 0; mt < 4; ++mt)
#pragma unroll
    for (int nt = 0; nt < 4; ++nt) acc2[mt][nt] = zf;

  __syncthreads();

  for (int j0 = 0; j0 < HID; j0 += TH) {
    // ---- GEMM1 + GEMM3: h = silu(x w1^T) * (x w3^T) for 16 cols/wave ----
    v8f a1[4], a3[4];
#pragma unroll
    for (int mt = 0; mt < 4; ++mt) { a1[mt] = zf; a3[mt] = zf; }
    const u16* b1p = w1e + (size_t)(j0 + wave * 16 + lr) * DIM;
    const u16* b3p = w3e + (size_t)(j0 + wave * 16 + lr) * DIM;
#pragma unroll 4
    for (int k0 = 0; k0 < DIM; k0 += 32) {
      const int ko = k0 + hsel * 8;
      v16bf b1 = load_frag(b1p + ko, b1p + ko + 16);
      v16bf b3 = load_frag(b3p + ko, b3p + ko + 16);
#pragma unroll
      for (int mt = 0; mt < 4; ++mt) {
        const u16* ap = xs + (mt * 16 + lr) * DIM + ko;
        v16bf a = load_frag(ap, ap + 16);
        a1[mt] = wmma_bf16(a, b1, a1[mt]);
        a3[mt] = wmma_bf16(a, b3, a3[mt]);
      }
    }
    __syncthreads();                 // prior GEMM2 readers of hs are done
#pragma unroll
    for (int mt = 0; mt < 4; ++mt) {
#pragma unroll
      for (int r = 0; r < 8; ++r) {
        float g = a1[mt][r];
        float hval = (g / (1.f + __expf(-g))) * a3[mt][r];  // silu * mul
        hs[(mt * 16 + hsel * 8 + r) * TH + wave * 16 + lr] = f2bf(hval);
      }
    }
    __syncthreads();

    // ---- GEMM2: out[64 x 64/wave] += h_chunk @ w2^T chunk ----
#pragma unroll
    for (int ks = 0; ks < TH; ks += 32) {
      const int ko = ks + hsel * 8;
      v16bf af[4];
#pragma unroll
      for (int mt = 0; mt < 4; ++mt) {
        const u16* ap = hs + (mt * 16 + lr) * TH + ko;
        af[mt] = load_frag(ap, ap + 16);
      }
#pragma unroll
      for (int nt = 0; nt < 4; ++nt) {
        const u16* bp = w2e + (size_t)(wave * 64 + nt * 16 + lr) * HID + j0 + ko;
        v16bf b = load_frag(bp, bp + 16);
#pragma unroll
        for (int mt = 0; mt < 4; ++mt)
          acc2[mt][nt] = wmma_bf16(af[mt], b, acc2[mt][nt]);
      }
    }
  }

  // Epilogue: gate-scale + atomic accumulate into output rows.
#pragma unroll
  for (int mt = 0; mt < 4; ++mt) {
#pragma unroll
    for (int r = 0; r < 8; ++r) {
      const int mloc = mt * 16 + hsel * 8 + r;
      const int tok = tok_s[mloc];
      if (tok < 0) continue;
      const float g = gate_s[mloc];
#pragma unroll
      for (int nt = 0; nt < 4; ++nt) {
        float* dst = out + (size_t)tok * DIM + wave * 64 + nt * 16 + lr;
        unsafeAtomicAdd(dst, g * acc2[mt][nt][r]);   // global_atomic_add_f32
      }
    }
  }
}

// ---------------------------------------------------------------- launch
extern "C" void kernel_launch(void* const* d_in, const int* in_sizes, int n_in,
                              void* d_out, int out_size, void* d_ws, size_t ws_size,
                              hipStream_t stream) {
  const float* x  = (const float*)d_in[0];
  const float* rw = (const float*)d_in[1];
  const float* w1 = (const float*)d_in[2];
  const float* w2 = (const float*)d_in[3];
  const float* w3 = (const float*)d_in[4];
  float* out = (float*)d_out;

  char* p = (char*)d_ws;
  int*   counts = (int*)p;              // 8 ints   (zeroed)
  int*   offs   = counts + 8;           // 9 ints
  int*   cursor = offs + 16;            // 8 ints
  float* usage  = (float*)(cursor + 8); // 8 floats (zeroed)
  p += 1024;
  u16* xb  = (u16*)p; p += (size_t)NTOK * DIM * 2;
  u16* w1b = (u16*)p; p += (size_t)NE * HID * DIM * 2;
  u16* w2b = (u16*)p; p += (size_t)NE * DIM * HID * 2;
  u16* w3b = (u16*)p; p += (size_t)NE * HID * DIM * 2;
  int*   topk_e = (int*)p;   p += (size_t)NTOK * TOPK * 4;
  float* topk_g = (float*)p; p += (size_t)NTOK * TOPK * 4;
  int*   gidx   = (int*)p;   p += (size_t)NTOK * TOPK * 4;
  float* ggate  = (float*)p; p += (size_t)NTOK * TOPK * 4;

  (void)hipMemsetAsync(d_out, 0, (size_t)out_size * sizeof(float), stream);
  (void)hipMemsetAsync(d_ws, 0, 1024, stream);

  { int n8 = NTOK * DIM / 8;
    f32_to_bf16_kernel<<<(n8 + 255) / 256, 256, 0, stream>>>(x, xb, n8); }
  { int n8 = NE * HID * DIM / 8;
    f32_to_bf16_kernel<<<(n8 + 255) / 256, 256, 0, stream>>>(w1, w1b, n8);
    f32_to_bf16_kernel<<<(n8 + 255) / 256, 256, 0, stream>>>(w2, w2b, n8);
    f32_to_bf16_kernel<<<(n8 + 255) / 256, 256, 0, stream>>>(w3, w3b, n8); }

  router_kernel<<<NTOK / 256, 256, 0, stream>>>(x, rw, topk_e, topk_g, counts, usage);
  finalize_router_kernel<<<1, 32, 0, stream>>>(counts, offs, cursor, usage,
                                               out + (size_t)NTOK * DIM);
  scatter_kernel<<<(NTOK * TOPK) / 256, 256, 0, stream>>>(topk_e, topk_g, cursor,
                                                          gidx, ggate);
  dim3 grid(NTOK / TM, NE);   // worst case: all tokens on one expert
  moe_ffn_kernel<<<grid, 256, 0, stream>>>(xb, w1b, w2b, w3b, gidx, ggate, offs, out);
}
